// PredLayer_66133906423975
// MI455X (gfx1250) — matrix operand order repaired and checked by reading, not compiled, not run on hardware
//
#include <hip/hip_runtime.h>
#include <math.h>

// Problem constants (match reference).
#define N_ROWS 4096
#define DIM    1024
#define VOCAB  32000
#define BM     128            // rows per workgroup tile
#define BV     128            // vocab cols per workgroup tile
#define NVB    (VOCAB / BV)   // 250 vocab blocks
#define NSTAGE 16             // K staged 64 at a time: 1024/64
#define SMOOTH 0.1f

// LDS staging: [row/col][k0..63] bf16 = 128 B data, padded row stride 144 B (36 dwords).
// 16-lane ds_load_b128 fragment reads start at distinct multiples of 4 dwords mod 64
// -> bank-conflict free.
#define STRB       144
#define TILE_BYTES (128 * STRB)              // 18432 per (x|w) buffer
#define XOFF(b)    ((b) * 2 * TILE_BYTES)
#define WOFF(b)    ((b) * 2 * TILE_BYTES + TILE_BYTES)
#define STAGE_SMEM (4 * TILE_BYTES)          // 73728 B (double-buffered x+w)
#define EPI_SMEM   (BM * 129 * 4)            // 66048 B epilogue tile (aliases staging)
#define SMEM_B     (STAGE_SMEM > EPI_SMEM ? STAGE_SMEM : EPI_SMEM)

typedef __attribute__((ext_vector_type(16))) __bf16 v16bf;
typedef __attribute__((ext_vector_type(8)))  __bf16 v8bf;
typedef __attribute__((ext_vector_type(8)))  float  v8f;

#if defined(__has_builtin) && __has_builtin(__builtin_amdgcn_s_wait_asynccnt)
#define WAIT_ASYNC(n) __builtin_amdgcn_s_wait_asynccnt((unsigned short)(n))
#else
#define WAIT_ASYNC(n) asm volatile("s_wait_asynccnt %0" ::"i"(n) : "memory")
#endif

// Byte offset of a generic pointer within the workgroup's LDS window.
__device__ __forceinline__ unsigned to_lds_off(const void* p) {
    return (unsigned)(unsigned long long)
        (__attribute__((address_space(3))) const void*)(p);
}

// Async global->LDS 16B copy, GVS mode: mem = SADDR64 + VADDR_I32; VDST = LDS byte addr.
__device__ __forceinline__ void async_copy16(unsigned long long gbase,
                                             unsigned goff, unsigned loff) {
    asm volatile("global_load_async_to_lds_b128 %0, %1, %2"
                 :
                 : "v"(loff), "v"(goff), "s"(gbase)
                 : "memory");
}

// fp32 -> bf16 round-to-nearest-even (bit-exact RNE)
__device__ __forceinline__ unsigned f2bf_rne(float f) {
    unsigned u = __float_as_uint(f);
    return (u + 0x7FFFu + ((u >> 16) & 1u)) >> 16;
}

// ---------------- Phase 0: convert x and W to bf16 (vectorized, bandwidth-bound) ----------------
__global__ void cvt_kernel(const float4* __restrict__ x4, const float4* __restrict__ W4,
                           uint2* __restrict__ xb4, uint2* __restrict__ wb4) {
    long long i      = (long long)blockIdx.x * blockDim.x + threadIdx.x;
    long long stride = (long long)gridDim.x * blockDim.x;
    const long long xcnt = (long long)N_ROWS * DIM / 4;
    const long long wcnt = (long long)VOCAB * DIM / 4;
    for (long long j = i; j < xcnt; j += stride) {
        float4 v = x4[j];
        xb4[j] = make_uint2(f2bf_rne(v.x) | (f2bf_rne(v.y) << 16),
                            f2bf_rne(v.z) | (f2bf_rne(v.w) << 16));
    }
    for (long long j = i; j < wcnt; j += stride) {
        float4 v = W4[j];
        wb4[j] = make_uint2(f2bf_rne(v.x) | (f2bf_rne(v.y) << 16),
                            f2bf_rne(v.z) | (f2bf_rne(v.w) << 16));
    }
}

// ---------------- Phase 1: fused GEMM tile + per-row softmax partials ----------------
// Grid: (VOCAB/BV, N_ROWS/BM), block 256 threads = 8 wave32 arranged 4(M) x 2(V).
// Each wave: 32 rows x 64 cols = 2 A-frags x 4 B-frags = 8 WMMA accumulators.
// K staged 64 at a time into LDS via double-buffered async copies (ASYNCcnt pipeline).
__global__ __launch_bounds__(256, 1)
void gemm_softmax_partial(const unsigned short* __restrict__ xb16,
                          const unsigned short* __restrict__ wb16,
                          const int* __restrict__ y,
                          float* __restrict__ partials,   // [N_ROWS][NVB][3]
                          float* __restrict__ tgt)        // [N_ROWS]
{
    __shared__ __align__(16) unsigned char smem[SMEM_B];

    const int vb   = blockIdx.x;
    const int mb   = blockIdx.y;
    const int tid  = threadIdx.x;
    const int w    = tid >> 5;
    const int lane = tid & 31;
    const int half = lane >> 4;
    const int l16  = lane & 15;
    const int wm   = w >> 1;    // 0..3 : row strip of 32
    const int wv   = w & 1;     // 0..1 : col strip of 64

    const unsigned long long gxb =
        (unsigned long long)(xb16 + (long long)(mb * BM) * DIM);
    const unsigned long long gwb =
        (unsigned long long)(wb16 + (long long)(vb * BV) * DIM);
    const unsigned ldsb = to_lds_off(smem);

    // Staging chunks: per stage per tile 128 rows x 128 B = 1024 x16B chunks; 4/thread/tile.
    unsigned goff[4], loff[4];
    #pragma unroll
    for (int j = 0; j < 4; ++j) {
        const int c   = tid + 256 * j;
        const int row = c >> 3, sub = c & 7;
        goff[j] = (unsigned)(row * (DIM * 2) + sub * 16);
        loff[j] = (unsigned)(row * STRB + sub * 16);
    }

    v8f acc[2][4];
    #pragma unroll
    for (int a = 0; a < 2; ++a)
        #pragma unroll
        for (int b = 0; b < 4; ++b) acc[a][b] = (v8f){};

    // Prologue: stage 0 into buffer 0 (8 async copies per thread).
    #pragma unroll
    for (int j = 0; j < 4; ++j) {
        async_copy16(gxb, goff[j], ldsb + XOFF(0) + loff[j]);
        async_copy16(gwb, goff[j], ldsb + WOFF(0) + loff[j]);
    }

    for (int st = 0; st < NSTAGE; ++st) {
        const int cur = st & 1;

        __syncthreads();   // everyone done reading buf[cur^1] (compute st-1)
        if (st + 1 < NSTAGE) {
            const int nb        = cur ^ 1;
            const unsigned stgb = (unsigned)((st + 1) * 128);  // stage K-offset in bytes
            #pragma unroll
            for (int j = 0; j < 4; ++j) {
                async_copy16(gxb, goff[j] + stgb, ldsb + XOFF(nb) + loff[j]);
                async_copy16(gwb, goff[j] + stgb, ldsb + WOFF(nb) + loff[j]);
            }
            WAIT_ASYNC(8);   // stage st's 8 copies done; st+1's in flight
        } else {
            WAIT_ASYNC(0);
        }
        __syncthreads();   // buf[cur] fully populated for all waves

        const unsigned char* xsb = smem + XOFF(cur);
        const unsigned char* wsb = smem + WOFF(cur);

        #pragma unroll
        for (int kc = 0; kc < 2; ++kc) {   // two K=32 slabs per stage
            // A fragments (16x32 bf16): lane(l16,half): K=half*16..+15 (contiguous).
            v16bf afr[2];
            #pragma unroll
            for (int fa = 0; fa < 2; ++fa) {
                const unsigned char* base =
                    xsb + (wm * 32 + fa * 16 + l16) * STRB + kc * 64;
                v8bf a0 = *(const v8bf*)(base + half * 32);
                v8bf a1 = *(const v8bf*)(base + 16 + half * 32);
                afr[fa] = __builtin_shufflevector(a0, a1,
                            0, 1, 2, 3, 4, 5, 6, 7, 8, 9, 10, 11, 12, 13, 14, 15);
            }
            #pragma unroll
            for (int fb = 0; fb < 4; ++fb) {
                const int col = wv * 64 + fb * 16 + l16;
                v16bf bfr = *(const v16bf*)(wsb + col * STRB + kc * 64 + half * 32);
                #pragma unroll
                for (int fa = 0; fa < 2; ++fa) {
                    acc[fa][fb] = __builtin_amdgcn_wmma_f32_16x16x32_bf16(
                                      false, afr[fa], false, bfr, (short)0, acc[fa][fb],
                                      false, false);
                }
            }
        }
    }

    // ------------- Epilogue: scatter C to LDS tile, then per-row partials -------------
    __syncthreads();   // staging region dead; safe to overwrite with fp32 tile
    float* tile = (float*)smem;
    #pragma unroll
    for (int fa = 0; fa < 2; ++fa) {
        #pragma unroll
        for (int fb = 0; fb < 4; ++fb) {
            #pragma unroll
            for (int i = 0; i < 8; ++i) {
                const int m = wm * 32 + fa * 16 + i + 8 * half;
                const int n = wv * 64 + fb * 16 + l16;
                tile[m * 129 + n] = acc[fa][fb][i];
            }
        }
    }
    __syncthreads();

    const int cbase = vb * BV;
    if (tid < BM) {
        const int r   = tid;
        const int row = mb * BM + r;
        const float* trow = &tile[r * 129];
        float m = -INFINITY, s = 0.0f;
        #pragma unroll 4
        for (int c = 0; c < BV; ++c) { float v = trow[c]; m = fmaxf(m, v); s += v; }
        float z = 0.0f;
        #pragma unroll 4
        for (int c = 0; c < BV; ++c) z += expf(trow[c] - m);
        float* p = partials + ((long long)row * NVB + vb) * 3;
        p[0] = m; p[1] = z; p[2] = s;
        int yc = y[row];
        if (yc >= cbase && yc < cbase + BV) tgt[row] = trow[yc - cbase];
    }
}

// ---------------- Phase 2: merge vocab-block partials per row ----------------
__global__ void row_merge(const float* __restrict__ partials, const float* __restrict__ tgt,
                          float* __restrict__ rowloss) {
    int row = blockIdx.x * blockDim.x + threadIdx.x;
    if (row >= N_ROWS) return;
    const float* p = partials + (long long)row * NVB * 3;
    float m = -INFINITY;
    for (int b = 0; b < NVB; ++b) m = fmaxf(m, p[b * 3 + 0]);
    float z = 0.0f, s = 0.0f;
    for (int b = 0; b < NVB; ++b) {
        z += p[b * 3 + 1] * expf(p[b * 3 + 0] - m);
        s += p[b * 3 + 2];
    }
    float lse = m + logf(z);
    rowloss[row] = lse - (1.0f - SMOOTH) * tgt[row] - (SMOOTH / (float)VOCAB) * s;
}

// ---------------- Phase 3: deterministic final mean ----------------
__global__ void final_sum(const float* __restrict__ rowloss, float* __restrict__ out) {
    __shared__ float red[256];
    float s = 0.0f;
    for (int i = threadIdx.x; i < N_ROWS; i += 256) s += rowloss[i];
    red[threadIdx.x] = s;
    __syncthreads();
    for (int off = 128; off > 0; off >>= 1) {
        if ((int)threadIdx.x < off) red[threadIdx.x] += red[threadIdx.x + off];
        __syncthreads();
    }
    if (threadIdx.x == 0) out[0] = red[0] / (float)N_ROWS;
}

extern "C" void kernel_launch(void* const* d_in, const int* in_sizes, int n_in,
                              void* d_out, int out_size, void* d_ws, size_t ws_size,
                              hipStream_t stream) {
    const float* x = (const float*)d_in[0];
    const float* W = (const float*)d_in[1];
    const int*   y = (const int*)d_in[2];

    char* ws = (char*)d_ws;
    unsigned short* xb = (unsigned short*)ws;                                   // 8 MB
    unsigned short* wb = (unsigned short*)(ws + (size_t)N_ROWS * DIM * 2);      // 64 MB
    float* partials = (float*)(ws + (size_t)N_ROWS * DIM * 2
                                  + (size_t)VOCAB * DIM * 2);                   // 12.3 MB
    float* tgt     = partials + (size_t)N_ROWS * NVB * 3;                       // 16 KB
    float* rowloss = tgt + N_ROWS;                                              // 16 KB

    cvt_kernel<<<2048, 256, 0, stream>>>((const float4*)x, (const float4*)W,
                                         (uint2*)xb, (uint2*)wb);

    dim3 grid(VOCAB / BV, N_ROWS / BM);   // 250 x 32
    gemm_softmax_partial<<<grid, 256, 0, stream>>>(xb, wb, y, partials, tgt);

    row_merge<<<N_ROWS / 256, 256, 0, stream>>>(partials, tgt, rowloss);
    final_sum<<<1, 256, 0, stream>>>(rowloss, (float*)d_out);
}
